// PLholonet_block_23854248362151
// MI455X (gfx1250) — matrix operand order repaired and checked by reading, not compiled, not run on hardware
//
#include <hip/hip_runtime.h>
#include <math.h>

// ---------------------------------------------------------------------------
// PLHoloNet forward for MI455X (gfx1250, wave32).
//   FFT2D   : LDS radix-2 256-pt kernels (row pass + strided col pass)
//   conv3x3 : implicit-GEMM via v_wmma_f32_16x16x32_bf16,
//             fragment-order LDS layouts (ds_load_b128 per half-fragment),
//             weights DMA'd to LDS by the Tensor Data Mover when available
//   rest    : float4-vectorized fused elementwise kernels (pure HBM streaming)
// ---------------------------------------------------------------------------

typedef __attribute__((ext_vector_type(16))) __bf16 v16bf;
typedef __attribute__((ext_vector_type(8)))  __bf16 v8bf;
typedef __attribute__((ext_vector_type(8)))  float  v8f;
typedef __attribute__((ext_vector_type(4)))  unsigned v4u;
typedef __attribute__((ext_vector_type(8)))  int      v8i;
typedef __attribute__((ext_vector_type(4)))  int      v4i;

#if defined(__has_builtin)
#if __has_builtin(__builtin_amdgcn_tensor_load_to_lds) && __has_builtin(__builtin_amdgcn_s_wait_tensorcnt)
#define ATHENA_TDM 1
#endif
#endif

#define Hh   256
#define Ww   256
#define Pp   65536            // H*W
#define Bb   4
#define Dd   32
#define BPn  (Bb*Pp)          // 262144  ([B,1,H,W] elements)
#define BDPn (Bb*Dd*Pp)       // 8388608 ([B,D,H,W] elements)

static __device__ __forceinline__ float lrelu01(float v) { return v > 0.f ? v : 0.01f * v; }

// ---------------------------------------------------------------------------
// 256-point radix-2 DIT FFT in LDS. One line per workgroup (256 threads).
// ---------------------------------------------------------------------------
__global__ __launch_bounds__(256)
void k_fft256(float2* __restrict__ d, int cols, int inverse) {
    __shared__ float2 sd[256];
    const int t = threadIdx.x;
    const unsigned blk = blockIdx.x;
    size_t base; int stride;
    if (cols) { base = (size_t)(blk >> 8) * Pp + (blk & 255u); stride = Ww; }
    else      { base = (size_t)blk * 256u;                     stride = 1; }

    const unsigned r = __brev((unsigned)t) >> 24;   // 8-bit bit-reversal
    sd[r] = d[base + (size_t)t * stride];
    __syncthreads();

    const float sgn = inverse ? 1.f : -1.f;
    for (int s = 0; s < 8; ++s) {
        const int half = 1 << s;
        if (t < 128) {
            const int j = t & (half - 1);
            const int k = ((t >> s) << (s + 1)) + j;
            const float ang = sgn * 3.14159265358979323846f * (float)j / (float)half;
            float sw, cw; __sincosf(ang, &sw, &cw);
            float2 u = sd[k];
            float2 v = sd[k + half];
            float2 vw = make_float2(v.x * cw - v.y * sw, v.x * sw + v.y * cw);
            sd[k]        = make_float2(u.x + vw.x, u.y + vw.y);
            sd[k + half] = make_float2(u.x - vw.x, u.y - vw.y);
        }
        __syncthreads();
    }
    float2 o = sd[t];
    if (inverse) { o.x *= (1.f / 256.f); o.y *= (1.f / 256.f); }
    d[base + (size_t)t * stride] = o;
}

// ---------------------------------------------------------------------------
// 3x3 SAME conv, 32->32 ch, implicit GEMM on WMMA bf16 (wave32).
// Workgroup = 8 waves; wave w computes row (by+w), 16 pixels x 32 couts.
// LDS layouts are fragment-ordered so each half-fragment is one ds_load_b128:
//   sIn[y][x][ci(pad 40)]      A half at &sIn[py][px][laneHi*8] / +16
//   sB [tap][frag][lane][e(24)] B half at &sB[tp][f][lane][0] / [8]
// Weights are staged to LDS by the Tensor Data Mover (one 36 KB descriptor,
// issued by wave 0, overlapped with VALU input-tile staging), then repacked
// to bf16 fragment order.
// ---------------------------------------------------------------------------
__global__ __launch_bounds__(256)
void k_conv3x3_wmma(const float* __restrict__ in, const float* __restrict__ w,
                    float* __restrict__ out) {
    __shared__ __align__(16) __bf16 sIn[10][18][40];   // 14.4 KB (pad: banks+align)
    __shared__ __align__(16) __bf16 sB[9][2][32][24];  // 27.6 KB fragment-order weights
#if defined(ATHENA_TDM)
    __shared__ __align__(16) float  sWstage[9216];     // raw f32 weights via TDM
#endif

    const int tid = threadIdx.x;
    const int bx = blockIdx.x * 16;      // output x tile
    const int by = blockIdx.y * 8;       // output y tile
    const int bb = blockIdx.z;           // batch
    const float* inB = in + (size_t)bb * Dd * Pp;

    __builtin_prefetch(inB + (size_t)by * Ww + bx, 0, 1);   // global_prefetch_b8

#if defined(ATHENA_TDM)
    if ((tid >> 5) == 0) {               // wave 0 issues the TDM DMA (EXEC-independent)
        const unsigned ldsAddr = (unsigned)(size_t)(&sWstage[0]);   // low 32 = LDS offset
        const unsigned long long ga = (unsigned long long)(size_t)w;
        // D# group0: count=1 | lds_addr | global_addr[56:0] | type=2
        v4u g0 = { 1u, ldsAddr, (unsigned)(ga & 0xFFFFFFFFull),
                   (unsigned)((ga >> 32) & 0x01FFFFFFull) | (2u << 30) };
        // D# group1: data_size=4B, tensor 9216x1, tile 9216x1, dim0 stride 9216
        v8i g1 = { (int)(2u << 16),
                   (int)((9216u & 0xFFFFu) << 16),   // tensor_dim0[15:0] @ [63:48]
                   (int)(1u << 16),                  // tensor_dim1 @ [95:80]
                   (int)(9216u << 16),               // tile_dim0 @ [127:112]
                   1,                                // tile_dim1 @ [143:128]
                   9216,                             // tensor_dim0_stride lo32
                   0, 0 };
        v4i gz = { 0, 0, 0, 0 };
#if __clang_major__ >= 23
        v8i gz8 = { 0, 0, 0, 0, 0, 0, 0, 0 };
        __builtin_amdgcn_tensor_load_to_lds(g0, g1, gz, gz, gz8, 0);
#else
        __builtin_amdgcn_tensor_load_to_lds(g0, g1, gz, gz, 0);
#endif
    }
#endif

    // stage input halo tile (zero-padded SAME borders), channel-contiguous
    for (int i = tid; i < 32 * 10 * 18; i += 256) {
        const int ci = i / 180;
        const int rr = i % 180;
        const int y  = rr / 18;
        const int x  = rr % 18;
        const int gy = by + y - 1, gx = bx + x - 1;
        float v = 0.f;
        if (gy >= 0 && gy < Hh && gx >= 0 && gx < Ww)
            v = inB[(size_t)ci * Pp + (size_t)gy * Ww + gx];
        sIn[y][x][ci] = (__bf16)v;
    }

#if defined(ATHENA_TDM)
    if ((tid >> 5) == 0) __builtin_amdgcn_s_wait_tensorcnt((short)0);
    __syncthreads();                     // sWstage visible to all waves
#endif

    // repack weights (cout,cin,tap) -> bf16 fragment order sB[tap][frag][lane][e]
    for (int i = tid; i < 9 * 2 * 32 * 16; i += 256) {
        const int e  = i & 15;
        const int l  = (i >> 4) & 31;
        const int f  = (i >> 9) & 1;
        const int tp = i >> 10;
        const int co = (l & 15) + f * 16;
        const int ci = (e & 7) + (l >> 4) * 8 + (e >> 3) * 16;
#if defined(ATHENA_TDM)
        sB[tp][f][l][e] = (__bf16)sWstage[(co * 32 + ci) * 9 + tp];
#else
        sB[tp][f][l][e] = (__bf16)w[(co * 32 + ci) * 9 + tp];
#endif
    }
    __syncthreads();

    const int wave   = tid >> 5;
    const int lane   = tid & 31;
    const int laneHi = lane >> 4;
    const int lrow   = lane & 15;

    v8f c0 = {};   // couts 0..15
    v8f c1 = {};   // couts 16..31

    for (int tp = 0; tp < 9; ++tp) {
        const int ky = tp / 3, kx = tp % 3;
        const int py = wave + ky;
        const int px = lrow + kx;

        // A frag: two aligned b128 LDS loads (channels {0..7,16..23} or {8..15,24..31})
        const v8bf alo = *(const v8bf*)&sIn[py][px][laneHi * 8];
        const v8bf ahi = *(const v8bf*)&sIn[py][px][16 + laneHi * 8];
        v16bf a;
        #pragma unroll
        for (int e = 0; e < 8; ++e) { a[e] = alo[e]; a[8 + e] = ahi[e]; }

        // B frags: two aligned b128 LDS loads each (fragment-order layout)
        const v8bf b0lo = *(const v8bf*)&sB[tp][0][lane][0];
        const v8bf b0hi = *(const v8bf*)&sB[tp][0][lane][8];
        const v8bf b1lo = *(const v8bf*)&sB[tp][1][lane][0];
        const v8bf b1hi = *(const v8bf*)&sB[tp][1][lane][8];
        v16bf b0, b1;
        #pragma unroll
        for (int e = 0; e < 8; ++e) {
            b0[e] = b0lo[e]; b0[8 + e] = b0hi[e];
            b1[e] = b1lo[e]; b1[8 + e] = b1hi[e];
        }
        c0 = __builtin_amdgcn_wmma_f32_16x16x32_bf16(false, a, false, b0, (short)0, c0, false, false);
        c1 = __builtin_amdgcn_wmma_f32_16x16x32_bf16(false, a, false, b1, (short)0, c1, false, false);
    }

    // D layout: lane<16 elem r -> (M=r, N=lane); lane>=16 -> (M=8+r, N=lane-16)
    // Per-lane the 8 elements are consecutive pixels -> global_store_b128 x2 per frag.
    float* outB = out + (size_t)bb * Dd * Pp + (size_t)(by + wave) * Ww + bx + (laneHi << 3);
    #pragma unroll
    for (int r = 0; r < 8; ++r) {
        outB[(size_t)lrow * Pp + r]        = c0[r];
        outB[(size_t)(lrow + 16) * Pp + r] = c1[r];
    }
}

// ---------------------------------------------------------------------------
// BatchNorm stats (training mode, biased var): one block per channel, f4 loads.
// stats[2c] = mean, stats[2c+1] = rsqrt(var + eps)
// ---------------------------------------------------------------------------
__global__ __launch_bounds__(256)
void k_bn_stats(const float* __restrict__ y, float* __restrict__ stats) {
    __shared__ float ss[256], ss2[256];
    const int c = blockIdx.x;
    const int tid = threadIdx.x;
    float s = 0.f, s2 = 0.f;
    for (int i = tid; i < (Bb * Pp) / 4; i += 256) {
        const int b  = i >> 14;                 // (i*4) >> 16
        const int p4 = i & 16383;
        const float4 v = ((const float4*)(y + ((size_t)b * Dd + c) * Pp))[p4];
        s  += v.x + v.y + v.z + v.w;
        s2 += v.x * v.x + v.y * v.y + v.z * v.z + v.w * v.w;
    }
    ss[tid] = s; ss2[tid] = s2;
    __syncthreads();
    for (int o = 128; o > 0; o >>= 1) {
        if (tid < o) { ss[tid] += ss[tid + o]; ss2[tid] += ss2[tid + o]; }
        __syncthreads();
    }
    if (tid == 0) {
        const float inv = 1.f / (float)(Bb * Pp);
        const float m = ss[0] * inv;
        const float var = ss2[0] * inv - m * m;
        stats[2 * c]     = m;
        stats[2 * c + 1] = rsqrtf(var + 1e-5f);
    }
}

// BN apply (+opt pre-act, residual, post-act, z_tilde subtraction), float4.
__global__ __launch_bounds__(256)
void k_bn_apply(const float4* __restrict__ y4, const float* __restrict__ gma,
                const float* __restrict__ bta, const float* __restrict__ stats,
                const float4* __restrict__ res4, const float4* __restrict__ zsub4,
                float4* __restrict__ out4, int preAct, int postAct) {
    const int i = blockIdx.x * 256 + threadIdx.x;
    if (i >= BDPn / 4) return;
    const int c = (i >> 14) & (Dd - 1);
    const float sc = stats[2 * c + 1] * gma[c];
    const float sh = bta[c] - stats[2 * c] * sc;
    float4 v = y4[i];
    v.x = v.x * sc + sh; v.y = v.y * sc + sh; v.z = v.z * sc + sh; v.w = v.w * sc + sh;
    if (preAct) { v.x = lrelu01(v.x); v.y = lrelu01(v.y); v.z = lrelu01(v.z); v.w = lrelu01(v.w); }
    if (res4) { const float4 r = res4[i]; v.x += r.x; v.y += r.y; v.z += r.z; v.w += r.w; }
    if (postAct) { v.x = lrelu01(v.x); v.y = lrelu01(v.y); v.z = lrelu01(v.z); v.w = lrelu01(v.w); }
    if (zsub4) { const float4 zt = zsub4[i]; v.x -= zt.x; v.y -= zt.y; v.z -= zt.z; v.w -= zt.w; }
    out4[i] = v;
}

// ---------------------------------------------------------------------------
// Elementwise kernels (float4 / 2-complex per lane)
// ---------------------------------------------------------------------------
__global__ __launch_bounds__(256)
void k_pack_sub_r2c(float4* __restrict__ c4, const float4* __restrict__ a4,
                    const float4* __restrict__ b4, int n4) {
    const int i = blockIdx.x * 256 + threadIdx.x;
    if (i >= n4) return;
    const float4 a = a4[i], b = b4[i];
    c4[2 * i]     = make_float4(a.x - b.x, 0.f, a.y - b.y, 0.f);
    c4[2 * i + 1] = make_float4(a.z - b.z, 0.f, a.w - b.w, 0.f);
}

__global__ __launch_bounds__(256)
void k_pack_r2c(float4* __restrict__ c4, const float4* __restrict__ a4, int n4) {
    const int i = blockIdx.x * 256 + threadIdx.x;
    if (i >= n4) return;
    const float4 a = a4[i];
    c4[2 * i]     = make_float4(a.x, 0.f, a.y, 0.f);
    c4[2 * i + 1] = make_float4(a.z, 0.f, a.w, 0.f);
}

// cA[b,d,p] = Fh[b,p] * conj(otf[b,d,p]) ; 2 complex per lane
__global__ __launch_bounds__(256)
void k_backproj(float4* __restrict__ cA4, const float4* __restrict__ Fh4,
                const float2* __restrict__ otr2, const float2* __restrict__ oti2) {
    const int i = blockIdx.x * 256 + threadIdx.x;
    if (i >= BDPn / 2) return;
    const int b = i >> 20;                       // plane = (2i)>>16 ; b = plane>>5
    const int p = (2 * i) & (Pp - 1);
    const float4 h = Fh4[(((size_t)b << 16) + p) >> 1];
    const float2 orr = otr2[i], oii = oti2[i];
    cA4[i] = make_float4(h.x * orr.x + h.y * oii.x, h.y * orr.x - h.x * oii.x,
                         h.z * orr.y + h.w * oii.y, h.w * orr.y - h.z * oii.y);
}

// rT = rho1*Re(cA) + rho2*(z - u2) ; 4 elements per lane
__global__ __launch_bounds__(256)
void k_ncombine(float4* __restrict__ rT4, const float4* __restrict__ cA4,
                const float4* __restrict__ z4, const float4* __restrict__ u24,
                const float* __restrict__ rho1, const float* __restrict__ rho2) {
    const int i = blockIdx.x * 256 + threadIdx.x;
    if (i >= BDPn / 4) return;
    const float r1 = rho1[0], r2 = rho2[0];
    const float4 a = cA4[2 * i], b = cA4[2 * i + 1];
    const float4 z = z4[i], u = u24[i];
    rT4[i] = make_float4(r1 * a.x + r2 * (z.x - u.x), r1 * a.z + r2 * (z.y - u.y),
                         r1 * b.x + r2 * (z.z - u.z), r1 * b.z + r2 * (z.w - u.w));
}

// cA /= (rho2 + rho1*|otf|^2) ; 2 complex per lane
__global__ __launch_bounds__(256)
void k_freqdiv(float4* __restrict__ cA4, const float2* __restrict__ otr2,
               const float2* __restrict__ oti2, const float* __restrict__ rho1,
               const float* __restrict__ rho2) {
    const int i = blockIdx.x * 256 + threadIdx.x;
    if (i >= BDPn / 2) return;
    const float r1 = rho1[0], r2 = rho2[0];
    const float2 orr = otr2[i], oii = oti2[i];
    const float i0 = 1.f / (r2 + r1 * (orr.x * orr.x + oii.x * oii.x));
    const float i1 = 1.f / (r2 + r1 * (orr.y * orr.y + oii.y * oii.y));
    float4 v = cA4[i];
    cA4[i] = make_float4(v.x * i0, v.y * i0, v.z * i1, v.w * i1);
}

// x = Re(cA); z_tilde = x + u2 ; 4 elements per lane
__global__ __launch_bounds__(256)
void k_x_ztilde(float4* __restrict__ x4, float4* __restrict__ zt4,
                const float4* __restrict__ cA4, const float4* __restrict__ u24) {
    const int i = blockIdx.x * 256 + threadIdx.x;
    if (i >= BDPn / 4) return;
    const float4 a = cA4[2 * i], b = cA4[2 * i + 1];
    const float4 x = make_float4(a.x, a.z, b.x, b.z);
    const float4 u = u24[i];
    x4[i]  = x;
    zt4[i] = make_float4(x.x + u.x, x.y + u.y, x.z + u.z, x.w + u.w);
}

// cB[b,p] = sum_d cA[b,d,p] * otf[b,d,p] ; 2 complex per lane
__global__ __launch_bounds__(256)
void k_fholo(float4* __restrict__ cB4, const float4* __restrict__ cA4,
             const float2* __restrict__ otr2, const float2* __restrict__ oti2) {
    const int i = blockIdx.x * 256 + threadIdx.x;
    if (i >= BPn / 2) return;
    const int b = i >> 15;
    const int p = (2 * i) & (Pp - 1);
    float4 acc = make_float4(0.f, 0.f, 0.f, 0.f);
    for (int d = 0; d < Dd; ++d) {
        const size_t idx = (((size_t)(b * Dd + d) << 16) + p) >> 1;
        const float4 v = cA4[idx];
        const float2 orr = otr2[idx], oii = oti2[idx];
        acc.x += v.x * orr.x - v.y * oii.x;
        acc.y += v.x * oii.x + v.y * orr.x;
        acc.z += v.z * orr.y - v.w * oii.y;
        acc.w += v.z * oii.y + v.w * orr.y;
    }
    cB4[i] = acc;
}

__global__ __launch_bounds__(256)
void k_absc(float2* __restrict__ fp2, const float4* __restrict__ c4, int n2) {
    const int i = blockIdx.x * 256 + threadIdx.x;
    if (i >= n2) return;
    const float4 v = c4[i];
    fp2[i] = make_float2(sqrtf(v.x * v.x + v.y * v.y), sqrtf(v.z * v.z + v.w * v.w));
}

// per-batch min/max of fp[b,:], one block per b, float4 loads
__global__ __launch_bounds__(256)
void k_minmax(const float* __restrict__ fp, float* __restrict__ mm) {
    __shared__ float smn[256], smx[256];
    const int b = blockIdx.x;
    const int tid = threadIdx.x;
    const float4* f4 = (const float4*)(fp + (size_t)b * Pp);
    float mn = 3.4e38f, mx = -3.4e38f;
    for (int i = tid; i < Pp / 4; i += 256) {
        const float4 v = f4[i];
        mn = fminf(mn, fminf(fminf(v.x, v.y), fminf(v.z, v.w)));
        mx = fmaxf(mx, fmaxf(fmaxf(v.x, v.y), fmaxf(v.z, v.w)));
    }
    smn[tid] = mn; smx[tid] = mx;
    __syncthreads();
    for (int o = 128; o > 0; o >>= 1) {
        if (tid < o) { smn[tid] = fminf(smn[tid], smn[tid + o]);
                       smx[tid] = fmaxf(smx[tid], smx[tid + o]); }
        __syncthreads();
    }
    if (tid == 0) { mm[2 * b] = smn[0]; mm[2 * b + 1] = smx[0]; }
}

__global__ __launch_bounds__(256)
void k_mmnorm(float4* __restrict__ fp4, const float* __restrict__ mm) {
    const int i = blockIdx.x * 256 + threadIdx.x;
    if (i >= BPn / 4) return;
    const int b = i >> 14;
    const float mn = mm[2 * b];
    const float inv = 1.f / (mm[2 * b + 1] - mn);
    float4 v = fp4[i];
    fp4[i] = make_float4((v.x - mn) * inv, (v.y - mn) * inv, (v.z - mn) * inv, (v.w - mn) * inv);
}

// Phi update: masked 30-step bisection (c = ALPHA/KFRAMES = 1) + u1 update.
__global__ __launch_bounds__(256)
void k_phi_bisect(float* __restrict__ phi_out, float* __restrict__ u1_out,
                  const float* __restrict__ fp, const float* __restrict__ u1,
                  const int* __restrict__ K1, const float* __restrict__ rho1p) {
    const int i = blockIdx.x * 256 + threadIdx.x;
    if (i >= BPn) return;
    const float rho1 = rho1p[0];
    const float pt = fp[i] + u1[i];        // phi_tilde
    const float K1f = (float)K1[i];
    float phin;
    if (K1f == 0.f) {
        phin = pt - 1.f / rho1;            // K0 = 1
    } else {
        const float K0 = 1.f - K1f;
        float pmin = 1e-5f, pmax = 100.f, pave = 0.5f * (pmin + pmax);
        bool done = false;
        for (int it = 0; it < 30; ++it) {
            const float t = (K0 - K1f / (__expf(pave) - 1.f)) + rho1 * (pave - pt);
            if (!done) {
                if (t > 0.f) pmin = pave;
                if (t < 0.f) pmax = pave;
                if (t == 0.f) done = true;
                if (!done) pave = 0.5f * (pmin + pmax);
            }
        }
        phin = pave;
    }
    phi_out[i] = phin;
    u1_out[i]  = u1[i] + phin - fp[i];
}

__global__ __launch_bounds__(256)
void k_softthr(float4* __restrict__ out4, const float4* __restrict__ h4,
               const float* __restrict__ thr) {
    const int i = blockIdx.x * 256 + threadIdx.x;
    if (i >= BDPn / 4) return;
    const float s = thr[0];
    const float4 v = h4[i];
    float4 o;
    float r;
    r = fmaxf(fabsf(v.x) - s, 0.f); o.x = v.x > 0.f ? r : -r;
    r = fmaxf(fabsf(v.y) - s, 0.f); o.y = v.y > 0.f ? r : -r;
    r = fmaxf(fabsf(v.z) - s, 0.f); o.z = v.z > 0.f ? r : -r;
    r = fmaxf(fabsf(v.w) - s, 0.f); o.w = v.w > 0.f ? r : -r;
    out4[i] = o;
}

__global__ __launch_bounds__(256)
void k_u2next(float4* __restrict__ u2o4, const float4* __restrict__ u24,
              const float4* __restrict__ zn4, const float4* __restrict__ x4) {
    const int i = blockIdx.x * 256 + threadIdx.x;
    if (i >= BDPn / 4) return;
    const float4 u = u24[i], zn = zn4[i], x = x4[i];
    u2o4[i] = make_float4(u.x + zn.x - x.x, u.y + zn.y - x.y,
                          u.z + zn.z - x.z, u.w + zn.w - x.w);
}

// ---------------------------------------------------------------------------
// Host-side orchestration
// ---------------------------------------------------------------------------
static inline void fft2(float2* d, int planes, int inverse, hipStream_t s) {
    k_fft256<<<planes * 256, 256, 0, s>>>(d, 0, inverse);   // rows
    k_fft256<<<planes * 256, 256, 0, s>>>(d, 1, inverse);   // cols
}

extern "C" void kernel_launch(void* const* d_in, const int* in_sizes, int n_in,
                              void* d_out, int out_size, void* d_ws, size_t ws_size,
                              hipStream_t stream) {
    (void)in_sizes; (void)n_in; (void)out_size; (void)ws_size;
    const float* phi  = (const float*)d_in[1];
    const float* z    = (const float*)d_in[2];
    const float* u1   = (const float*)d_in[3];
    const float* u2   = (const float*)d_in[4];
    const float* otr  = (const float*)d_in[5];
    const float* oti  = (const float*)d_in[6];
    const int*   K1   = (const int*)d_in[7];
    const float* rho1 = (const float*)d_in[8];
    const float* rho2 = (const float*)d_in[9];
    const float* sthr = (const float*)d_in[10];
    const float* w1a = (const float*)d_in[11], *g1a = (const float*)d_in[12], *b1a = (const float*)d_in[13];
    const float* w1b = (const float*)d_in[14], *g1b = (const float*)d_in[15], *b1b = (const float*)d_in[16];
    const float* w2a = (const float*)d_in[17], *g2a = (const float*)d_in[18], *b2a = (const float*)d_in[19];
    const float* w2b = (const float*)d_in[20], *g2b = (const float*)d_in[21], *b2b = (const float*)d_in[22];

    float* out     = (float*)d_out;
    float* out_x   = out;                  // [B,D,H,W]
    float* out_phi = out_x   + BDPn;       // [B,1,H,W]
    float* out_z   = out_phi + BPn;        // [B,D,H,W]
    float* out_u1  = out_z   + BDPn;       // [B,1,H,W]
    float* out_u2  = out_u1  + BPn;        // [B,D,H,W]
    float* out_sym = out_u2  + BDPn;       // [B,D,H,W]

    float*  W   = (float*)d_ws;
    float2* cA  = (float2*)W;                         // [B,D,H,W] complex
    float*  rT1 = W + 2 * (size_t)BDPn;               // scratch reals
    float*  rT2 = rT1 + BDPn;                         // z_tilde
    float*  rT3 = rT2 + BDPn;                         // h1
    float*  rT4 = rT3 + BDPn;                         // h1_thr / cbl temp
    float*  rT5 = rT4 + BDPn;                         // conv output scratch
    float2* cB  = (float2*)(rT5 + BDPn);              // [B,1,H,W] complex
    float*  fpb = (float*)cB + 2 * (size_t)BPn;       // forward-proj [B,1,H,W]
    float*  st  = fpb + BPn;                          // BN stats (64) + minmax (8)
    float*  mm  = st + 64;

    const int gD4 = BDPn / 4 / 256;   // 8192
    const int gD2 = BDPn / 2 / 256;   // 16384
    const int gP  = BPn / 256;        // 1024
    const int gP4 = BPn / 4 / 256;    // 256
    const int gP2 = BPn / 2 / 256;    // 512
    const dim3 gConv(Ww / 16, Hh / 8, Bb);

    // ================= X update =================
    k_pack_sub_r2c<<<gP4, 256, 0, stream>>>((float4*)cB, (const float4*)phi, (const float4*)u1, BPn / 4);
    fft2(cB, Bb, 0, stream);                                       // Fh
    k_backproj<<<gD2, 256, 0, stream>>>((float4*)cA, (const float4*)cB, (const float2*)otr, (const float2*)oti);
    fft2(cA, Bb * Dd, 1, stream);                                  // back_proj
    k_ncombine<<<gD4, 256, 0, stream>>>((float4*)rT1, (const float4*)cA, (const float4*)z, (const float4*)u2, rho1, rho2);
    k_pack_r2c<<<gD4, 256, 0, stream>>>((float4*)cA, (const float4*)rT1, BDPn / 4);
    fft2(cA, Bb * Dd, 0, stream);
    k_freqdiv<<<gD2, 256, 0, stream>>>((float4*)cA, (const float2*)otr, (const float2*)oti, rho1, rho2);
    fft2(cA, Bb * Dd, 1, stream);
    k_x_ztilde<<<gD4, 256, 0, stream>>>((float4*)out_x, (float4*)rT2, (const float4*)cA, (const float4*)u2);

    // ============ forward_proj(x, otf) ============
    k_pack_r2c<<<gD4, 256, 0, stream>>>((float4*)cA, (const float4*)out_x, BDPn / 4);
    fft2(cA, Bb * Dd, 0, stream);
    k_fholo<<<gP2, 256, 0, stream>>>((float4*)cB, (const float4*)cA, (const float2*)otr, (const float2*)oti);
    fft2(cB, Bb, 1, stream);
    k_absc<<<gP2, 256, 0, stream>>>((float2*)fpb, (const float4*)cB, BPn / 2);
    k_minmax<<<Bb, 256, 0, stream>>>(fpb, mm);
    k_mmnorm<<<gP4, 256, 0, stream>>>((float4*)fpb, mm);

    // ============ Phi + u1 update ============
    k_phi_bisect<<<gP, 256, 0, stream>>>(out_phi, out_u1, fpb, u1, K1, rho1);

    // ============ Z update: denoiser ============
    // resblock1(z_tilde)
    k_conv3x3_wmma<<<gConv, 256, 0, stream>>>(rT2, w1a, rT5);
    k_bn_stats<<<Dd, 256, 0, stream>>>(rT5, st);
    k_bn_apply<<<gD4, 256, 0, stream>>>((const float4*)rT5, g1a, b1a, st, nullptr, nullptr, (float4*)rT4, 1, 0);
    k_conv3x3_wmma<<<gConv, 256, 0, stream>>>(rT4, w1b, rT5);
    k_bn_stats<<<Dd, 256, 0, stream>>>(rT5, st);
    k_bn_apply<<<gD4, 256, 0, stream>>>((const float4*)rT5, g1b, b1b, st, (const float4*)rT2, nullptr, (float4*)rT3, 0, 1);

    k_softthr<<<gD4, 256, 0, stream>>>((float4*)rT4, (const float4*)rT3, sthr);  // h1_thr

    // resblock2(h1_thr) -> z_next
    k_conv3x3_wmma<<<gConv, 256, 0, stream>>>(rT4, w2a, rT5);
    k_bn_stats<<<Dd, 256, 0, stream>>>(rT5, st);
    k_bn_apply<<<gD4, 256, 0, stream>>>((const float4*)rT5, g2a, b2a, st, nullptr, nullptr, (float4*)rT1, 1, 0);
    k_conv3x3_wmma<<<gConv, 256, 0, stream>>>(rT1, w2b, rT5);
    k_bn_stats<<<Dd, 256, 0, stream>>>(rT5, st);
    k_bn_apply<<<gD4, 256, 0, stream>>>((const float4*)rT5, g2b, b2b, st, (const float4*)rT4, nullptr, (float4*)out_z, 0, 1);

    // resblock2(h1) -> x_fb ; sym = x_fb - z_tilde (fused)
    k_conv3x3_wmma<<<gConv, 256, 0, stream>>>(rT3, w2a, rT5);
    k_bn_stats<<<Dd, 256, 0, stream>>>(rT5, st);
    k_bn_apply<<<gD4, 256, 0, stream>>>((const float4*)rT5, g2a, b2a, st, nullptr, nullptr, (float4*)rT1, 1, 0);
    k_conv3x3_wmma<<<gConv, 256, 0, stream>>>(rT1, w2b, rT5);
    k_bn_stats<<<Dd, 256, 0, stream>>>(rT5, st);
    k_bn_apply<<<gD4, 256, 0, stream>>>((const float4*)rT5, g2b, b2b, st, (const float4*)rT3, (const float4*)rT2, (float4*)out_sym, 0, 1);

    // ============ u2 update ============
    k_u2next<<<gD4, 256, 0, stream>>>((float4*)out_u2, (const float4*)u2, (const float4*)out_z, (const float4*)out_x);
}
